// ModulatedConv2d_9028021256349
// MI455X (gfx1250) — matrix operand order repaired and checked by reading, not compiled, not run on hardware
//
#include <hip/hip_runtime.h>
#include <hip/hip_bf16.h>

typedef __attribute__((ext_vector_type(16))) _Float16 v16h;
typedef __attribute__((ext_vector_type(8)))  _Float16 v8h;
typedef __attribute__((ext_vector_type(8)))  float    v8f;

constexpr int BATCH = 8;
constexpr int FIN   = 256;
constexpr int FOUT  = 256;
constexpr int H     = 128;
constexpr int W     = 128;
constexpr int LAT   = 512;

// ---------------------------------------------------------------------------
// Kernel 1: modulation scales  s[b][fi] = latent[b] . mlp_w[fi] * sqrt(2/512) + mlp_b[fi]
// ---------------------------------------------------------------------------
__global__ void mod_scale_kernel(const float* __restrict__ latent,
                                 const float* __restrict__ mlp_w,
                                 const float* __restrict__ mlp_b,
                                 float* __restrict__ sbuf) {
    const int b  = blockIdx.x;
    const int fi = threadIdx.x;
    const float lscale = 0.0625f;               // sqrt(2/512) exactly
    const float* lv = latent + b * LAT;
    const float* wv = mlp_w + fi * LAT;
    float acc = 0.0f;
    #pragma unroll 4
    for (int d = 0; d < LAT; ++d) acc += lv[d] * wv[d];
    sbuf[b * FIN + fi] = acc * lscale + mlp_b[fi];
}

// ---------------------------------------------------------------------------
// Kernel 2: demodulated f16 weights, layout [b][tap][fib][fo][fi%32]
//   (one contiguous 256x32 A-chunk per (b,tap,fib) K-step of the conv kernel)
// ---------------------------------------------------------------------------
__global__ void demod_weights_kernel(const float* __restrict__ weight,
                                     const float* __restrict__ sbuf,
                                     _Float16* __restrict__ wmod) {
    const int fo = blockIdx.x;
    const int b  = blockIdx.y;
    const int fi = threadIdx.x;
    const float cscale = 0.02946278254943948f;  // sqrt(2/(256*9))

    const float sv = sbuf[b * FIN + fi] * cscale;
    const float* wp = weight + (fo * FIN + fi) * 9;
    float wv[9];
    float ss = 0.0f;
    #pragma unroll
    for (int t = 0; t < 9; ++t) { wv[t] = wp[t] * sv; ss += wv[t] * wv[t]; }

    __shared__ float red[FIN];
    red[fi] = ss;
    __syncthreads();
    for (int off = FIN / 2; off > 0; off >>= 1) {
        if (fi < off) red[fi] += red[fi + off];
        __syncthreads();
    }
    const float dem = rsqrtf(red[0] + 1e-5f);

    const int fib = fi >> 5, fil = fi & 31;
    #pragma unroll
    for (int t = 0; t < 9; ++t)
        wmod[(size_t)(((b * 9 + t) * 8 + fib) * FOUT + fo) * 32 + fil] =
            (_Float16)(wv[t] * dem);
}

// ---------------------------------------------------------------------------
// Kernel 3: implicit-GEMM grouped conv via v_wmma_f32_16x16x32_f16
//   grid: (256 pixel-tiles, BATCH); block: 256 threads = 8 waves
//   tile: 256 fo x 64 px; each wave: 32(fo) x 64(px) = 8 WMMAs/K-step
//   K loop: 72 steps (3x3 taps x 8 blocks of 32 channels), double-buffered LDS
// ---------------------------------------------------------------------------
__global__ __launch_bounds__(256)
void modconv_wmma_kernel(const float* __restrict__ input,
                         const _Float16* __restrict__ wmod,
                         const float* __restrict__ bias,
                         float* __restrict__ out) {
    const int tile  = blockIdx.x;            // 0..255
    const int b     = blockIdx.y;
    const int h     = tile >> 1;             // output row
    const int wbase = (tile & 1) * 64;       // output col base

    const int t    = threadIdx.x;
    const int lane = t & 31;
    const int wave = t >> 5;                 // wave32: wave covers fo rows 32*wave..+31

    // Padded rows (40 halfs = 80B): conflict-free, 16B-aligned chunks. Ping-pong.
    __shared__ _Float16 As[2][256 * 40];     // [fo][k]   20KB each
    __shared__ _Float16 Bs[2][64 * 40];      // [pixel][k] 5KB each

    v8f acc[2][4] = {};                      // [mi][ni] 16x16 f32 tiles

    // staging roles: A: thread t copies one contiguous 64B run (row t, 32 halfs)
    //                B: 64 px x 4 chunks of 8 k
    const int bn = t & 63, bkq = t >> 6;

    // fragment offsets (ISA 16-bit A 16x32 / B 32x16 layouts)
    const int ar = lane & 15, ahi = lane >> 4;
    const int aoff = (32 * wave + ar) * 40 + ahi * 8;          // +16*40 for mi=1
    const int boff = (lane & 15) * 40 + (lane >> 4) * 8;       // +16*40 per ni

    auto stage = [&](int step, int buf) {
        const int kh  = step / 24;
        const int rem = step - kh * 24;
        const int kw  = rem >> 3;
        const int fib = rem & 7;
        // ---- A: contiguous 256x32 f16 chunk; thread t copies 32 halfs at t*32
        const _Float16* wsrc =
            wmod + (size_t)(((b * 9 + (kh * 3 + kw)) * 8 + fib) * FOUT) * 32 + t * 32;
        _Float16* adst = &As[buf][t * 40];
        #pragma unroll
        for (int q = 0; q < 4; ++q)
            *(v8h*)(adst + q * 8) = *(const v8h*)(wsrc + q * 8);
        // ---- B: 32ch x 64px input slab, f32->f16, zero-padded halo
        const int hrow = h + kh - 1;
        const int col  = wbase + bn + kw - 1;
        const bool ok  = ((unsigned)hrow < (unsigned)H) && ((unsigned)col < (unsigned)W);
        const int hc   = ((unsigned)hrow < (unsigned)H) ? hrow : 0;
        const int cc   = ((unsigned)col  < (unsigned)W) ? col  : 0;
        const float* isrc =
            input + ((size_t)(b * FIN + fib * 32 + bkq * 8) * H + hc) * W + cc;
        __builtin_prefetch(isrc + (size_t)32 * H * W, 0, 1);  // next channel block -> L2
        v8h bv;
        #pragma unroll
        for (int j = 0; j < 8; ++j) {
            float v = isrc[(size_t)j * H * W];
            bv[j] = (_Float16)(ok ? v : 0.0f);
        }
        *(v8h*)&Bs[buf][bn * 40 + bkq * 8] = bv;
    };

    stage(0, 0);
    #pragma unroll 1
    for (int step = 0; step < 72; ++step) {
        __syncthreads();                       // staged data visible; prev reads done
        if (step < 71) stage(step + 1, (step + 1) & 1);
        const int cur = step & 1;
        const _Float16* Ab = As[cur];
        const _Float16* Bb = Bs[cur];
        union U { v16h v; v8h h2[2]; };
        U a[2], bf[4];
        #pragma unroll
        for (int mi = 0; mi < 2; ++mi) {
            a[mi].h2[0] = *(const v8h*)(Ab + aoff + mi * 16 * 40);
            a[mi].h2[1] = *(const v8h*)(Ab + aoff + mi * 16 * 40 + 16);
        }
        #pragma unroll
        for (int ni = 0; ni < 4; ++ni) {
            bf[ni].h2[0] = *(const v8h*)(Bb + boff + ni * 16 * 40);
            bf[ni].h2[1] = *(const v8h*)(Bb + boff + ni * 16 * 40 + 16);
        }
        #pragma unroll
        for (int mi = 0; mi < 2; ++mi)
            #pragma unroll
            for (int ni = 0; ni < 4; ++ni)
                acc[mi][ni] = __builtin_amdgcn_wmma_f32_16x16x32_f16(
                    false, a[mi].v, false, bf[ni].v, (short)0, acc[mi][ni], false, false);
    }

    // C layout: lanes 0-15 N=lane, M=v; lanes 16-31 N=lane-16, M=v+8
    const int mb = (lane >> 4) * 8;
    #pragma unroll
    for (int mi = 0; mi < 2; ++mi) {
        #pragma unroll
        for (int v = 0; v < 8; ++v) {
            const int fo = 32 * wave + 16 * mi + mb + v;
            const float bv = bias[fo];
            const size_t o = ((size_t)(b * FOUT + fo) * H + h) * W + wbase + (lane & 15);
            #pragma unroll
            for (int ni = 0; ni < 4; ++ni)
                out[o + 16 * ni] = acc[mi][ni][v] + bv;
        }
    }
}

// ---------------------------------------------------------------------------
extern "C" void kernel_launch(void* const* d_in, const int* in_sizes, int n_in,
                              void* d_out, int out_size, void* d_ws, size_t ws_size,
                              hipStream_t stream) {
    const float* input  = (const float*)d_in[0];   // (8,256,128,128)
    const float* latent = (const float*)d_in[1];   // (8,512)
    const float* weight = (const float*)d_in[2];   // (256,256,3,3)
    const float* bias   = (const float*)d_in[3];   // (1,256,1,1)
    const float* mlp_w  = (const float*)d_in[4];   // (256,512)
    const float* mlp_b  = (const float*)d_in[5];   // (256,)
    float* out = (float*)d_out;                    // (8,256,128,128)

    float*    sbuf = (float*)d_ws;                          // 8*256 f32 = 8KB
    _Float16* wmod = (_Float16*)((char*)d_ws + 8192);       // 8*9*8*256*32 f16 ≈ 9.4MB

    mod_scale_kernel<<<dim3(BATCH), dim3(FIN), 0, stream>>>(latent, mlp_w, mlp_b, sbuf);
    demod_weights_kernel<<<dim3(FOUT, BATCH), dim3(FIN), 0, stream>>>(weight, sbuf, wmod);
    modconv_wmma_kernel<<<dim3((H * W) / 64, BATCH), dim3(256), 0, stream>>>(
        input, wmod, bias, out);
}
